// Cluster_78683800862855
// MI455X (gfx1250) — matrix operand order repaired and checked by reading, not compiled, not run on hardware
//
#include <hip/hip_runtime.h>

// ---------------------------------------------------------------------------
// Cluster kernel for MI455X (gfx1250): euclidean cdist via bf16x3 WMMA
//   dists = sqrt(max(x2 + c2 - 2*x.c, 1e-12))
//   outputs: centers copy [K*D], rep_ids [K] (argmin over rows per center),
//            loss = sum over rows of min-over-centers distance
// Round 2: double-buffered GLOBAL_LOAD_ASYNC_TO_LDS staging of the B chunk
// (centers) overlapped with WMMA compute; ASYNCcnt-tracked, no VGPR bounce.
// ---------------------------------------------------------------------------

typedef __attribute__((ext_vector_type(16))) __bf16 v16bf;
typedef __attribute__((ext_vector_type(8)))  float  v8f;
typedef __attribute__((ext_vector_type(8)))  unsigned int u32x8;

#define DDIM   128   // feature dim
#define WAVES  8     // waves per block (256 threads, wave32)
#define ROWS_PER_WG 128   // 8 waves * 16 rows
#define CCOLS  32    // columns (centers) per LDS chunk (double-buffered)
#define CT     2     // 16-wide column tiles per chunk
#define KC     4     // 32-wide K chunks (DDIM/32)
#define MAXK   512
#define CHUNK_BYTES (CCOLS * DDIM * 2)  // 8192 bytes per hi/lo matrix

__device__ __forceinline__ unsigned long long uMin64(unsigned long long a,
                                                     unsigned long long b) {
  return a < b ? a : b;
}

// split 16 fp32 values (one fragment, contiguous k order) into truncation-based
// bf16 hi/lo packed words; accumulate sum of squares.
__device__ __forceinline__ void splitPack16(const float* f, u32x8& hw, u32x8& lw,
                                            float& x2) {
#pragma unroll
  for (int i = 0; i < 8; ++i) {
    float x0 = f[2 * i], x1 = f[2 * i + 1];
    unsigned int b0 = __float_as_uint(x0), b1 = __float_as_uint(x1);
    unsigned int h0 = b0 & 0xffff0000u, h1 = b1 & 0xffff0000u;
    float l0 = x0 - __uint_as_float(h0);
    float l1 = x1 - __uint_as_float(h1);
    hw[i] = (h0 >> 16) | h1;
    lw[i] = (__float_as_uint(l0) >> 16) | (__float_as_uint(l1) & 0xffff0000u);
    x2 = fmaf(x0, x0, x2);
    x2 = fmaf(x1, x1, x2);
  }
}

// Async global->LDS copy of one thread's share (2 x 16B of each 8KB matrix).
// IOFFSET applies to both LDS and memory address, so one base pair suffices.
__device__ __forceinline__ void stageAsync(const unsigned short* gh,
                                           const unsigned short* gl,
                                           const unsigned short* sh,
                                           const unsigned short* sl, int tid) {
  unsigned long long gH = (unsigned long long)gh + (unsigned long long)tid * 16;
  unsigned long long gL = (unsigned long long)gl + (unsigned long long)tid * 16;
  unsigned sH = (unsigned)(unsigned long long)sh + (unsigned)tid * 16;
  unsigned sL = (unsigned)(unsigned long long)sl + (unsigned)tid * 16;
  asm volatile(
      "global_load_async_to_lds_b128 %0, %1, off\n\t"
      "global_load_async_to_lds_b128 %0, %1, off offset:4096\n\t"
      "global_load_async_to_lds_b128 %2, %3, off\n\t"
      "global_load_async_to_lds_b128 %2, %3, off offset:4096"
      :
      : "v"(sH), "v"(gH), "v"(sL), "v"(gL)
      : "memory");
}

// ---------------------------------------------------------------------------
// Prep: centers fp32 -> bf16 hi/lo (row-major [k][d]) + c2[k]; copy to d_out.
// ---------------------------------------------------------------------------
__global__ void __launch_bounds__(128) prep_centers(
    const float* __restrict__ centers, unsigned short* __restrict__ chi,
    unsigned short* __restrict__ clo, float* __restrict__ c2g,
    float* __restrict__ outCenters) {
  __shared__ float red[4];
  const int k = blockIdx.x;
  const int d = threadIdx.x;
  const float x = centers[k * DDIM + d];
  outCenters[k * DDIM + d] = x;
  unsigned int b = __float_as_uint(x);
  unsigned int h = b & 0xffff0000u;
  float l = x - __uint_as_float(h);
  chi[k * DDIM + d] = (unsigned short)(h >> 16);
  clo[k * DDIM + d] = (unsigned short)(__float_as_uint(l) >> 16);
  float s = x * x;
#pragma unroll
  for (int o = 16; o > 0; o >>= 1) s += __shfl_xor(s, o, 32);
  if ((threadIdx.x & 31) == 0) red[threadIdx.x >> 5] = s;
  __syncthreads();
  if (threadIdx.x == 0) c2g[k] = red[0] + red[1] + red[2] + red[3];
}

__global__ void init_keys(unsigned long long* __restrict__ keys, int K) {
  int i = blockIdx.x * blockDim.x + threadIdx.x;
  if (i < K) keys[i] = ~0ULL;
}

// ---------------------------------------------------------------------------
// Main kernel: persistent blocks; each iteration handles 128 rows x all K cols.
// A (16 rows x 128, hi+lo bf16) register-resident per wave; B chunks streamed
// through double-buffered LDS via async loads.
// ---------------------------------------------------------------------------
__global__ void __launch_bounds__(256) cluster_main(
    const float* __restrict__ embs, const unsigned short* __restrict__ chi,
    const unsigned short* __restrict__ clo, const float* __restrict__ c2g,
    unsigned long long* __restrict__ keys, float* __restrict__ partials,
    int N, int K, int numGroups) {
  __shared__ __align__(16) unsigned short lbh[2][CCOLS * DDIM];
  __shared__ __align__(16) unsigned short lbl[2][CCOLS * DDIM];
  __shared__ unsigned long long colBest[MAXK];
  __shared__ float waveSum[WAVES];
  __shared__ float lossAcc;

  const int tid = threadIdx.x;
  const int lane = tid & 31;
  const int wave = tid >> 5;
  const int hh = lane >> 4;   // half-wave index (ISA fragment layout)
  const int lm = lane & 15;   // row (A) / column (B,C) within 16-tile

  for (int i = tid; i < K; i += 256) colBest[i] = ~0ULL;
  if (tid == 0) lossAcc = 0.f;
  __syncthreads();

  const int nChunks = K / CCOLS;

  for (int grp = blockIdx.x; grp < numGroups; grp += gridDim.x) {
    const int rowBase = grp * ROWS_PER_WG + wave * 16;

    // ---- load + convert A fragments (register resident for this iteration)
    v16bf a_hi[KC], a_lo[KC];
    float x2part = 0.f;
    const int rClamp = min(rowBase + lm, N - 1);
    const float* rp = embs + (size_t)rClamp * DDIM;
#pragma unroll
    for (int kc = 0; kc < KC; ++kc) {
      // element e of v16: K = kc*32 + 16*hh + e (16 contiguous)
      const float* b0 = rp + kc * 32 + 16 * hh;
      float4 q0 = *(const float4*)(b0);
      float4 q1 = *(const float4*)(b0 + 4);
      float4 q2 = *(const float4*)(b0 + 8);
      float4 q3 = *(const float4*)(b0 + 12);
      float f[16] = {q0.x, q0.y, q0.z, q0.w, q1.x, q1.y, q1.z, q1.w,
                     q2.x, q2.y, q2.z, q2.w, q3.x, q3.y, q3.z, q3.w};
      u32x8 hw, lw;
      splitPack16(f, hw, lw, x2part);
      a_hi[kc] = __builtin_bit_cast(v16bf, hw);
      a_lo[kc] = __builtin_bit_cast(v16bf, lw);
    }
    float x2full = x2part + __shfl_xor(x2part, 16, 32);  // ||row lm||^2
    if (rowBase + lm >= N) x2full = 1e30f;               // invalidate tail rows
    float x2j[8];
#pragma unroll
    for (int j = 0; j < 8; ++j) x2j[j] = __shfl(x2full, j + 8 * hh, 32);

    float rowmin[8];
#pragma unroll
    for (int j = 0; j < 8; ++j) rowmin[j] = 3.0e38f;

    // prologue: async-stage chunk 0 into buffer 0
    stageAsync(chi, clo, &lbh[0][0], &lbl[0][0], tid);
    int buf = 0;

    for (int ch = 0; ch < nChunks; ++ch) {
      const int cbase = ch * CCOLS;
      if (ch + 1 < nChunks) {  // prefetch next chunk into the other buffer
        const size_t nb = (size_t)(ch + 1) * CCOLS * DDIM;
        stageAsync(chi + nb, clo + nb, &lbh[buf ^ 1][0], &lbl[buf ^ 1][0], tid);
        asm volatile("s_wait_asynccnt 0x4" ::: "memory");  // chunk ch landed
      } else {
        asm volatile("s_wait_asynccnt 0x0" ::: "memory");
      }
      __syncthreads();  // all waves' async writes for chunk ch visible

      float c2v[CT];
#pragma unroll
      for (int ct = 0; ct < CT; ++ct) c2v[ct] = c2g[cbase + ct * 16 + lm];

      v8f acc[CT];
#pragma unroll
      for (int ct = 0; ct < CT; ++ct) acc[ct] = (v8f){0, 0, 0, 0, 0, 0, 0, 0};

#pragma unroll
      for (int kc = 0; kc < KC; ++kc) {
#pragma unroll
        for (int ct = 0; ct < CT; ++ct) {
          // B fragment: element e -> K = kc*32 + 16*hh + e (16 contiguous)
          const int col = ct * 16 + lm;
          const uint4* ph =
              (const uint4*)&lbh[buf][col * DDIM + kc * 32 + 16 * hh];
          const uint4* pl =
              (const uint4*)&lbl[buf][col * DDIM + kc * 32 + 16 * hh];
          uint4 h0 = ph[0], h1 = ph[1];
          uint4 l0 = pl[0], l1 = pl[1];
          u32x8 bhw = {h0.x, h0.y, h0.z, h0.w, h1.x, h1.y, h1.z, h1.w};
          u32x8 blw = {l0.x, l0.y, l0.z, l0.w, l1.x, l1.y, l1.z, l1.w};
          v16bf bh = __builtin_bit_cast(v16bf, bhw);
          v16bf bl = __builtin_bit_cast(v16bf, blw);
          // bf16x3: x.c ~= xh.ch + xh.cl + xl.ch
          acc[ct] = __builtin_amdgcn_wmma_f32_16x16x32_bf16(
              false, a_hi[kc], false, bh, (short)0, acc[ct], false, false);
          acc[ct] = __builtin_amdgcn_wmma_f32_16x16x32_bf16(
              false, a_hi[kc], false, bl, (short)0, acc[ct], false, false);
          acc[ct] = __builtin_amdgcn_wmma_f32_16x16x32_bf16(
              false, a_lo[kc], false, bh, (short)0, acc[ct], false, false);
        }
      }

      // ---- epilogue: d2 = max(x2 + c2 - 2*xc, 1e-12)
      float dmin[8];
#pragma unroll
      for (int j = 0; j < 8; ++j) dmin[j] = 3.0e38f;
#pragma unroll
      for (int ct = 0; ct < CT; ++ct) {
        unsigned long long bestkey = ~0ULL;
#pragma unroll
        for (int j = 0; j < 8; ++j) {
          // C layout: VGPR j, lane -> row = j + 8*hh, col = lm
          float xc = acc[ct][j];
          float d2 = fmaxf(fmaf(-2.f, xc, x2j[j] + c2v[ct]), 1e-12f);
          dmin[j] = fminf(dmin[j], d2);
          unsigned long long key =
              ((unsigned long long)__float_as_uint(d2) << 32) |
              (unsigned int)(rowBase + j + 8 * hh);
          bestkey = uMin64(bestkey, key);
        }
        bestkey = uMin64(bestkey, __shfl_xor(bestkey, 16, 32));
        if (hh == 0)
          atomicMin(&colBest[cbase + ct * 16 + lm], bestkey);  // ds_min_u64
      }
#pragma unroll
      for (int j = 0; j < 8; ++j) {  // min over the 16 columns of this half
        float v = dmin[j];
        v = fminf(v, __shfl_xor(v, 1, 32));
        v = fminf(v, __shfl_xor(v, 2, 32));
        v = fminf(v, __shfl_xor(v, 4, 32));
        v = fminf(v, __shfl_xor(v, 8, 32));
        rowmin[j] = fminf(rowmin[j], v);
      }
      __syncthreads();  // buf consumed; next iteration may overwrite it
      buf ^= 1;
    }  // chunks

    // ---- loss contribution of this iteration's 128 rows
    float wl = 0.f;
    if (lm == 0) {
#pragma unroll
      for (int j = 0; j < 8; ++j) {
        int r = rowBase + j + 8 * hh;
        if (r < N) wl += sqrtf(rowmin[j]);
      }
    }
    wl += __shfl_xor(wl, 16, 32);
    if (lane == 0) waveSum[wave] = wl;
    __syncthreads();
    if (tid == 0) {
      float s = 0.f;
      for (int w = 0; w < WAVES; ++w) s += waveSum[w];
      lossAcc += s;
    }
    __syncthreads();  // waveSum/lossAcc settled before next iteration
  }

  __syncthreads();
  for (int i = tid; i < K; i += 256)
    atomicMin(&keys[i], colBest[i]);  // global_atomic_min_u64
  if (tid == 0) partials[blockIdx.x] = lossAcc;
}

// ---------------------------------------------------------------------------
// Finalize: rep_ids from packed keys; deterministic serial loss sum.
// ---------------------------------------------------------------------------
__global__ void __launch_bounds__(256) finalize(
    const unsigned long long* __restrict__ keys,
    const float* __restrict__ partials, float* __restrict__ outRep,
    float* __restrict__ outLoss, int K, int G) {
  for (int i = threadIdx.x; i < K; i += blockDim.x)
    outRep[i] = (float)(unsigned int)(keys[i] & 0xffffffffu);
  if (threadIdx.x == 0) {
    float s = 0.f;
    for (int g = 0; g < G; ++g) s += partials[g];
    *outLoss = s;
  }
}

extern "C" void kernel_launch(void* const* d_in, const int* in_sizes, int n_in,
                              void* d_out, int out_size, void* d_ws,
                              size_t ws_size, hipStream_t stream) {
  const float* embs = (const float*)d_in[0];
  const float* centers = (const float*)d_in[1];
  const int N = in_sizes[0] / DDIM;
  const int K = in_sizes[1] / DDIM;

  float* out = (float*)d_out;
  float* outCenters = out;                 // K*D
  float* outRep = out + (size_t)K * DDIM;  // K
  float* outLoss = outRep + K;             // 1

  char* ws = (char*)d_ws;
  auto carve = [&](size_t bytes) -> char* {
    char* p = ws;
    ws += (bytes + 255) & ~(size_t)255;
    return p;
  };
  unsigned short* chi = (unsigned short*)carve((size_t)K * DDIM * 2);
  unsigned short* clo = (unsigned short*)carve((size_t)K * DDIM * 2);
  float* c2g = (float*)carve((size_t)K * 4);
  unsigned long long* keys = (unsigned long long*)carve((size_t)K * 8);
  const int G = 512;
  float* partials = (float*)carve((size_t)G * 4);

  prep_centers<<<K, 128, 0, stream>>>(centers, chi, clo, c2g, outCenters);
  init_keys<<<(K + 255) / 256, 256, 0, stream>>>(keys, K);
  const int numGroups = (N + ROWS_PER_WG - 1) / ROWS_PER_WG;
  cluster_main<<<G, 256, 0, stream>>>(embs, chi, clo, c2g, keys, partials, N, K,
                                      numGroups);
  finalize<<<1, 256, 0, stream>>>(keys, partials, outRep, outLoss, K, G);
}
